// Code2Vec_22557168239480
// MI455X (gfx1250) — compile-verified, matrix operand order checked by reading, not run on hardware
//
#include <hip/hip_runtime.h>
#include <hip/hip_bf16.h>
#include <math.h>

// ---------------------------------------------------------------------------
// CDNA5 WMMA (wave32): V_WMMA_F32_16X16X32_BF16
//   A: 16x32 bf16 (8 VGPRs/lane), B: 32x16 bf16 (8 VGPRs/lane), C/D: 16x16 f32
// Per-lane operand layout (ISA 7.12.2): two contiguous 8-element K runs,
//   run0 @ k = kk + (lane>>4)*8, run1 @ run0 + 16; row/col = lane & 15.
// ---------------------------------------------------------------------------
typedef __attribute__((ext_vector_type(16))) __bf16 v16bf;
typedef __attribute__((ext_vector_type(8)))  __bf16 v8bf;
typedef __attribute__((ext_vector_type(8)))  float  v8f;

#define B_DIM   1024
#define L_DIM   200
#define M_DIM   (B_DIM * L_DIM)   // 204800 rows, multiple of 32
#define E_DIM   128
#define C_DIM   384                // = 3*E
#define OUT_DIM 10000              // = 625 * 16

__device__ __forceinline__ v8bf cvt8_f32_bf16(const float* __restrict__ p) {
    float4 x = *reinterpret_cast<const float4*>(p);
    float4 y = *reinterpret_cast<const float4*>(p + 4);
    v8bf r;
    r[0] = (__bf16)x.x; r[1] = (__bf16)x.y; r[2] = (__bf16)x.z; r[3] = (__bf16)x.w;
    r[4] = (__bf16)y.x; r[5] = (__bf16)y.y; r[6] = (__bf16)y.z; r[7] = (__bf16)y.w;
    return r;
}

__device__ __forceinline__ v8bf load8_bf16(const __bf16* __restrict__ p) {
    return *reinterpret_cast<const v8bf*>(p);   // one global_load_b128
}

__device__ __forceinline__ v16bf cat16(v8bf lo, v8bf hi) {
    return __builtin_shufflevector(lo, hi, 0,1,2,3,4,5,6,7,8,9,10,11,12,13,14,15);
}

__device__ __forceinline__ v16bf loadB16(const __bf16* __restrict__ p) {
    return cat16(load8_bf16(p), load8_bf16(p + 16));   // runs at k and k+16
}

// ---------------------------------------------------------------------------
// Kernel 0: f32 -> bf16 weight pre-conversion (vectorized, memory bound)
// ---------------------------------------------------------------------------
__global__ __launch_bounds__(256)
void cvt_weights_kernel(const float* __restrict__ in, __bf16* __restrict__ out, int n4) {
    int i = blockIdx.x * blockDim.x + threadIdx.x;     // processes 4 floats
    if (i >= n4) return;
    float4 x = reinterpret_cast<const float4*>(in)[i];
    __bf16* o = out + (size_t)i * 4;
    o[0] = (__bf16)x.x; o[1] = (__bf16)x.y; o[2] = (__bf16)x.z; o[3] = (__bf16)x.w;
}

// ---------------------------------------------------------------------------
// Kernel 1: fused gather + FC GEMM (bf16 WMMA) + tanh + h store + scores.
// Block = 128 (4 waves) handles TWO 16-row M tiles (32 rows).
// Wave owns 6 N-tiles (96 output channels); B fragments double-buffered.
// ---------------------------------------------------------------------------
__global__ __launch_bounds__(128)
void fc_tanh_score_kernel(const int*    __restrict__ starts,
                          const int*    __restrict__ paths,
                          const int*    __restrict__ ends,
                          const float*  __restrict__ node_emb,
                          const float*  __restrict__ path_emb,
                          const __bf16* __restrict__ Wfc_bf,   // [384,384] bf16
                          const float*  __restrict__ a_vec,
                          __bf16*       __restrict__ h_out,    // [M,384] bf16
                          float*        __restrict__ scores) { // [M] f32
    __shared__ float s_scores[32];
    const int tid   = threadIdx.x;
    const int wave  = tid >> 5;
    const int lane  = tid & 31;
    const int lm    = lane & 15;
    const int khalf = lane >> 4;
    const int mt0   = blockIdx.x * 2;          // first of two 16-row tiles
    const int m0    = mt0 * 16 + lm;
    const int m1    = m0 + 16;

    if (tid < 32) s_scores[tid] = 0.0f;
    __syncthreads();

    const int is0 = starts[m0], ip0 = paths[m0], ie0 = ends[m0];
    const int is1 = starts[m1], ip1 = paths[m1], ie1 = ends[m1];

    v8f acc0[6], acc1[6];
    const v8f zero = {0.f, 0.f, 0.f, 0.f, 0.f, 0.f, 0.f, 0.f};
#pragma unroll
    for (int t = 0; t < 6; ++t) { acc0[t] = zero; acc1[t] = zero; }

    for (int seg = 0; seg < 3; ++seg) {
        // each 32-wide K chunk lies fully inside one embedding segment
        const float* b0 = (seg == 0) ? node_emb + (size_t)is0 * E_DIM
                        : (seg == 1) ? path_emb + (size_t)ip0 * E_DIM
                                     : node_emb + (size_t)ie0 * E_DIM;
        const float* b1 = (seg == 0) ? node_emb + (size_t)is1 * E_DIM
                        : (seg == 1) ? path_emb + (size_t)ip1 * E_DIM
                                     : node_emb + (size_t)ie1 * E_DIM;
        for (int kc = 0; kc < 4; ++kc) {
            const int k0 = kc * 32 + khalf * 8;      // lane's first 8-K run
            const int kw = seg * 128 + k0;           // K offset into W row
            v16bf af0 = cat16(cvt8_f32_bf16(b0 + k0), cvt8_f32_bf16(b0 + k0 + 16));
            v16bf af1 = cat16(cvt8_f32_bf16(b1 + k0), cvt8_f32_bf16(b1 + k0 + 16));

            const __bf16* wbase = Wfc_bf + (size_t)(wave * 96 + lm) * C_DIM + kw;
            v16bf bf[2];
            bf[0] = loadB16(wbase);                  // t = 0 fragment
#pragma unroll
            for (int t = 0; t < 6; ++t) {
                if (t + 1 < 6)                       // prefetch next B fragment
                    bf[(t + 1) & 1] = loadB16(wbase + (size_t)(t + 1) * 16 * C_DIM);
                acc0[t] = __builtin_amdgcn_wmma_f32_16x16x32_bf16(
                              false, af0, false, bf[t & 1], (short)0, acc0[t], false, false);
                acc1[t] = __builtin_amdgcn_wmma_f32_16x16x32_bf16(
                              false, af1, false, bf[t & 1], (short)0, acc1[t], false, false);
            }
        }
    }

    // epilogue: tanh, bf16 store of h, score partials (lane holds one column)
    float a_vals[6];
#pragma unroll
    for (int t = 0; t < 6; ++t) a_vals[t] = a_vec[(wave * 6 + t) * 16 + lm];

#pragma unroll
    for (int tile = 0; tile < 2; ++tile) {
        float srow[8];
#pragma unroll
        for (int v = 0; v < 8; ++v) srow[v] = 0.0f;
#pragma unroll
        for (int t = 0; t < 6; ++t) {
            const int col = (wave * 6 + t) * 16 + lm;
#pragma unroll
            for (int v = 0; v < 8; ++v) {
                const float hv = tanhf(tile ? acc1[t][v] : acc0[t][v]);
                const int  row = (mt0 + tile) * 16 + v + 8 * khalf;
                h_out[(size_t)row * C_DIM + col] = (__bf16)hv;
                srow[v] += hv * a_vals[t];
            }
        }
#pragma unroll
        for (int v = 0; v < 8; ++v) {          // reduce over 16 cols in lane group
            float s = srow[v];
            s += __shfl_xor(s, 1, 32);
            s += __shfl_xor(s, 2, 32);
            s += __shfl_xor(s, 4, 32);
            s += __shfl_xor(s, 8, 32);
            if (lm == 0) atomicAdd(&s_scores[tile * 16 + v + 8 * khalf], s);
        }
    }
    __syncthreads();
    if (tid < 32) scores[mt0 * 16 + tid] = s_scores[tid];
}

// ---------------------------------------------------------------------------
// Kernel 2: softmax over L per batch, in place (scores -> attn).
// ---------------------------------------------------------------------------
__global__ __launch_bounds__(256)
void softmax_kernel(float* __restrict__ scores) {
    __shared__ float red[256];
    const int b = blockIdx.x, tid = threadIdx.x;
    float v = (tid < L_DIM) ? scores[b * L_DIM + tid] : -INFINITY;
    red[tid] = v; __syncthreads();
    for (int s = 128; s > 0; s >>= 1) {
        if (tid < s) red[tid] = fmaxf(red[tid], red[tid + s]);
        __syncthreads();
    }
    const float mx = red[0]; __syncthreads();
    const float e = (tid < L_DIM) ? __expf(v - mx) : 0.0f;
    red[tid] = e; __syncthreads();
    for (int s = 128; s > 0; s >>= 1) {
        if (tid < s) red[tid] += red[tid + s];
        __syncthreads();
    }
    const float inv = 1.0f / red[0];
    if (tid < L_DIM) scores[b * L_DIM + tid] = e * inv;
}

// ---------------------------------------------------------------------------
// Kernel 3: code_vectors[b,c] = sum_l attn[b,l] * h[b,l,c]  (memory bound)
// ---------------------------------------------------------------------------
__global__ __launch_bounds__(384)
void code_vec_kernel(const float*  __restrict__ attn,
                     const __bf16* __restrict__ h,
                     __bf16*       __restrict__ cvb) {
    const int b = blockIdx.x, c = threadIdx.x;
    const __bf16* hb = h + (size_t)b * L_DIM * C_DIM + c;
    const float*  ab = attn + (size_t)b * L_DIM;
    float acc = 0.0f;
    for (int l = 0; l < L_DIM; ++l)
        acc += ab[l] * (float)hb[(size_t)l * C_DIM];
    cvb[(size_t)b * C_DIM + c] = (__bf16)acc;
}

// ---------------------------------------------------------------------------
// Kernel 4: out = cv @ W_out^T + b_out via bf16 WMMA.
// grid = (ceil(625/8), 32); block = 256 (8 waves, 1 N-tile per wave,
// TWO M tiles per wave for 2x W_out reuse). All operands pre-converted bf16.
// ---------------------------------------------------------------------------
__global__ __launch_bounds__(256)
void out_gemm_kernel(const __bf16* __restrict__ cvb,      // [1024,384] bf16
                     const __bf16* __restrict__ Wout_bf,  // [10000,384] bf16
                     const float*  __restrict__ b_out,
                     float*        __restrict__ out) {    // [1024,10000] f32
    const int tid   = threadIdx.x;
    const int wave  = tid >> 5;
    const int lane  = tid & 31;
    const int lm    = lane & 15;
    const int khalf = lane >> 4;
    const int ntile = blockIdx.x * 8 + wave;
    if (ntile >= OUT_DIM / 16) return;                // uniform per wave
    const int mt0   = blockIdx.y * 2;
    const int ncol  = ntile * 16 + lm;

    const __bf16* ap0 = cvb + (size_t)(mt0 * 16 + lm) * C_DIM;
    const __bf16* ap1 = ap0 + (size_t)16 * C_DIM;
    const __bf16* wp  = Wout_bf + (size_t)ncol * C_DIM;

    v8f acc0 = {0.f,0.f,0.f,0.f,0.f,0.f,0.f,0.f};
    v8f acc1 = acc0;
    for (int kk = 0; kk < C_DIM; kk += 32) {
        const int k0 = kk + khalf * 8;
        v16bf bfrag = loadB16(wp + k0);
        v16bf af0   = loadB16(ap0 + k0);
        v16bf af1   = loadB16(ap1 + k0);
        acc0 = __builtin_amdgcn_wmma_f32_16x16x32_bf16(
                   false, af0, false, bfrag, (short)0, acc0, false, false);
        acc1 = __builtin_amdgcn_wmma_f32_16x16x32_bf16(
                   false, af1, false, bfrag, (short)0, acc1, false, false);
    }
    const float bias = b_out[ncol];
#pragma unroll
    for (int v = 0; v < 8; ++v) {
        const int r0 = mt0 * 16 + v + 8 * khalf;
        out[(size_t)r0 * OUT_DIM + ncol]        = acc0[v] + bias;
        out[(size_t)(r0 + 16) * OUT_DIM + ncol] = acc1[v] + bias;
    }
}

// ---------------------------------------------------------------------------
extern "C" void kernel_launch(void* const* d_in, const int* in_sizes, int n_in,
                              void* d_out, int out_size, void* d_ws, size_t ws_size,
                              hipStream_t stream) {
    const int*   starts   = (const int*)  d_in[0];
    const int*   paths    = (const int*)  d_in[1];
    const int*   ends     = (const int*)  d_in[2];
    const float* node_emb = (const float*)d_in[3];
    const float* path_emb = (const float*)d_in[4];
    const float* W_fc     = (const float*)d_in[5];
    const float* a_vec    = (const float*)d_in[6];
    const float* W_out    = (const float*)d_in[7];
    const float* b_out    = (const float*)d_in[8];
    float*       out      = (float*)d_out;

    // workspace layout
    char* ws = (char*)d_ws;
    __bf16* h_ws    = (__bf16*)ws;  ws += (size_t)M_DIM * C_DIM * sizeof(__bf16);
    float*  scores  = (float*)ws;   ws += (size_t)M_DIM * sizeof(float);
    __bf16* cvb     = (__bf16*)ws;  ws += (size_t)B_DIM * C_DIM * sizeof(__bf16);
    __bf16* wfc_bf  = (__bf16*)ws;  ws += (size_t)C_DIM * C_DIM * sizeof(__bf16);
    __bf16* wout_bf = (__bf16*)ws;
    (void)ws_size; (void)in_sizes; (void)n_in; (void)out_size;

    // weight pre-conversion (removes f32->bf16 VALU + per-load waits from GEMMs)
    {
        int n4 = (C_DIM * C_DIM) / 4;
        cvt_weights_kernel<<<(n4 + 255) / 256, 256, 0, stream>>>(W_fc, wfc_bf, n4);
        int m4 = (OUT_DIM * C_DIM) / 4;
        cvt_weights_kernel<<<(m4 + 255) / 256, 256, 0, stream>>>(W_out, wout_bf, m4);
    }

    fc_tanh_score_kernel<<<M_DIM / 32, 128, 0, stream>>>(
        starts, paths, ends, node_emb, path_emb, wfc_bf, a_vec, h_ws, scores);

    softmax_kernel<<<B_DIM, 256, 0, stream>>>(scores);

    code_vec_kernel<<<B_DIM, C_DIM, 0, stream>>>(scores, h_ws, cvb);

    dim3 grid4((OUT_DIM / 16 + 7) / 8, B_DIM / 32);
    out_gemm_kernel<<<grid4, 256, 0, stream>>>(cvb, wout_bf, b_out, out);
}
